// CardEncoder_46445776338948
// MI455X (gfx1250) — matrix operand order repaired and checked by reading, compiled, not run on hardware
//
#include <hip/hip_runtime.h>

// ---------------------------------------------------------------------------
// CardEncoder for MI455X (gfx1250, wave32, WMMA f32_16x16x32_f16)
//
//  ref:  emb = table[x]; bi-LSTM(HID=32, mask_zero) over L=16;
//        out = tanh(concat_states[2048,4096] @ Wd[4096,64] + bd)
//
//  kernel 0 (prep):  swizzle Wk/Wr/Wd into WMMA B-fragment order (f16) in ws
//  kernel 1 (lstm):  1 wave == 16 sequences; x-proj + recurrence fully fused,
//                    512 v_wmma per wave, h transposed via LDS each step
//  kernel 2 (head):  state[f16] @ WdF via WMMA, tanh, f32 out
//
//  ws layout: [0,8K)=WkF  [8K,16K)=WrF  [16K,528K)=WdF  [1M,17M)=state f16
// ---------------------------------------------------------------------------

typedef _Float16 v16h __attribute__((ext_vector_type(16)));
typedef _Float16 v8h  __attribute__((ext_vector_type(8)));
typedef float    v8f  __attribute__((ext_vector_type(8)));

#define WKF_OFF   0
#define WRF_OFF   8192
#define WDF_OFF   16384
#define STATE_OFF (1u << 20)

__device__ __forceinline__ float fexp2(float x) { return __builtin_amdgcn_exp2f(x); }
__device__ __forceinline__ float frcp(float x)  { return __builtin_amdgcn_rcpf(x); }
__device__ __forceinline__ float fsig(float x)  { return frcp(1.0f + fexp2(-1.44269504f * x)); }
__device__ __forceinline__ float ftanh(float x) { return 1.0f - 2.0f * frcp(1.0f + fexp2(2.88539008f * x)); }

__device__ __forceinline__ v16h cat8(v8h lo, v8h hi) {
  return __builtin_shufflevector(lo, hi, 0,1,2,3,4,5,6,7,8,9,10,11,12,13,14,15);
}

// ---------------------------------------------------------------------------
// Kernel 0: swizzle weights into B-fragment order.
// B frag (32x16 f16): lane l (<16) half h -> K=h<8 ? h : 8+h,      N=16j+(l&15)
//                     lane l (>=16) half h -> K=h<8 ? 8+h : 16+h,  N=16j+(l&15)
// WkF/WrF: halves at ((j*32+l)*16+h).  WdF: halves at (((s*4+j)*32+l)*16+h),
// k = 32*s + (h<8 ? h : 8+h) + (l<16 ? 0 : 8).
// ---------------------------------------------------------------------------
__global__ void prep_kernel(const float* __restrict__ Wk, const float* __restrict__ Wr,
                            const float* __restrict__ Wd,
                            _Float16* __restrict__ wkf, _Float16* __restrict__ wrf,
                            _Float16* __restrict__ wdf) {
  int idx = blockIdx.x * 256 + threadIdx.x;
  if (idx < 8192) {                         // Wk / Wr fragments (4096 each)
    int i = idx & 4095;
    int h = i & 15, l = (i >> 4) & 31, j = i >> 9;
    int col = j * 16 + (l & 15);
    int k   = ((h < 8) ? h : 8 + h) + ((l < 16) ? 0 : 8);
    float v = (idx < 4096) ? Wk[k * 128 + col] : Wr[k * 128 + col];
    ((idx < 4096) ? wkf : wrf)[i] = (_Float16)v;
  } else {                                  // Wd fragments (262144)
    int i = idx - 8192;
    if (i < 262144) {
      int h = i & 15, l = (i >> 4) & 31, j = (i >> 9) & 3, s = i >> 11;
      int col = j * 16 + (l & 15);
      int k   = 32 * s + ((h < 8) ? h : 8 + h) + ((l < 16) ? 0 : 8);
      wdf[i] = (_Float16)Wd[k * 64 + col];
    }
  }
}

// ---------------------------------------------------------------------------
// Kernel 1: fused bidirectional LSTM.  2 waves / block, 1 wave == 1 tile of 16
// sequences.  8192 tiles total -> 4096 blocks x 64 threads (exact, EXEC full).
// ---------------------------------------------------------------------------
template<bool FWD>
__device__ __forceinline__ void run_dir(const _Float16* __restrict__ ebw,  // [t][lane][16] halves
                                        const _Float16* __restrict__ wkf,  // [j][lane][16]
                                        const _Float16* __restrict__ wrf,
                                        _Float16* __restrict__ hb,         // [seq][32] halves
                                        const unsigned (&um)[8],
                                        const float (&bv)[8],
                                        int lane,
                                        _Float16* __restrict__ st) {       // global: this tile+dir
  const int  col = lane & 15;
  const bool hi  = lane >= 16;
  const int  sof = hi ? 8 : 0;

  v8f  c0 = {}, c1 = {}, h0 = {}, h1 = {};
  v16h hfrag = {};

#pragma unroll
  for (int step = 0; step < 16; ++step) {
    const int t = FWD ? step : 15 - step;

    // A-fragment of E_t: per-lane 32B block, no cross-lane traffic
    const _Float16* ep = ebw + (t * 32 + lane) * 16;
    v16h ef = cat8(*(const v8h*)ep, *(const v8h*)(ep + 8));

    v8f z[8];
#pragma unroll
    for (int j = 0; j < 8; ++j) {
      v8f acc;
#pragma unroll
      for (int r = 0; r < 8; ++r) acc[r] = bv[j];
      const _Float16* kp = wkf + (j * 32 + lane) * 16;
      const _Float16* rp = wrf + (j * 32 + lane) * 16;
      v16h bk = cat8(*(const v8h*)kp, *(const v8h*)(kp + 8));
      v16h br = cat8(*(const v8h*)rp, *(const v8h*)(rp + 8));
      acc = __builtin_amdgcn_wmma_f32_16x16x32_f16(false, ef,    false, bk, (short)0, acc, false, false);
      acc = __builtin_amdgcn_wmma_f32_16x16x32_f16(false, hfrag, false, br, (short)0, acc, false, false);
      z[j] = acc;
    }

    // gates in C-layout: element r <-> seq (r+sof), chunk0 = hid col, chunk1 = hid 16+col
#pragma unroll
    for (int r = 0; r < 8; ++r) {
      float i0 = fsig(z[0][r]), i1 = fsig(z[1][r]);
      float f0 = fsig(z[2][r]), f1 = fsig(z[3][r]);
      float g0 = ftanh(z[4][r]), g1 = ftanh(z[5][r]);
      float o0 = fsig(z[6][r]), o1 = fsig(z[7][r]);
      float cn0 = f0 * c0[r] + i0 * g0;
      float cn1 = f1 * c1[r] + i1 * g1;
      float hn0 = o0 * ftanh(cn0);
      float hn1 = o1 * ftanh(cn1);
      bool  m   = (um[r] >> t) & 1u;           // masked step: state unchanged
      c0[r] = m ? cn0 : c0[r];
      c1[r] = m ? cn1 : c1[r];
      h0[r] = m ? hn0 : h0[r];
      h1[r] = m ? hn1 : h1[r];
      hb[(r + sof) * 32 + col]      = (_Float16)h0[r];
      hb[(r + sof) * 32 + 16 + col] = (_Float16)h1[r];
    }

    // C-layout -> A-layout transpose of h via LDS (same-wave, in-order DS)
    asm volatile("s_wait_dscnt 0" ::: "memory");
    const _Float16* hp = hb + (lane & 15) * 32 + sof;
    hfrag = cat8(*(const v8h*)hp, *(const v8h*)(hp + 16));
  }

  // hb now holds the final masked h [16][32] f16 row-major; coalesced write-out
  asm volatile("s_wait_dscnt 0" ::: "memory");
#pragma unroll
  for (int q = 0; q < 2; ++q) {
    int idx = lane + 32 * q;                  // 0..63
    int row = idx >> 2, seg = idx & 3;        // 4 x 8-half segments per row
    v8h v = *(const v8h*)(hb + row * 32 + seg * 8);
    *(v8h*)(st + row * 64 + seg * 8) = v;
  }
}

__global__ __launch_bounds__(64) void lstm_kernel(const int* __restrict__ x,
                                                  const float* __restrict__ etab,
                                                  const float* __restrict__ bias,
                                                  const _Float16* __restrict__ wkf_g,
                                                  const _Float16* __restrict__ wrf_g,
                                                  _Float16* __restrict__ state) {
  __shared__ __align__(16) _Float16 wkf[8 * 32 * 16];
  __shared__ __align__(16) _Float16 wrf[8 * 32 * 16];
  __shared__ __align__(16) _Float16 ebuf[2][16 * 32 * 16];
  __shared__ __align__(16) _Float16 hbuf[2][16 * 32];
  __shared__ unsigned mbuf[2][16];

  const int tid  = threadIdx.x;
  const int w    = tid >> 5;
  const int lane = tid & 31;

  // stage pre-swizzled weight fragments (shared by both waves)
  for (int i = tid; i < 512; i += 64) {
    ((float4*)wkf)[i] = ((const float4*)wkf_g)[i];
    ((float4*)wrf)[i] = ((const float4*)wrf_g)[i];
  }

  const int  tile = blockIdx.x * 2 + w;
  const int  seq  = lane & 15;
  const bool hi   = lane >= 16;
  const int  k0   = hi ? 8 : 0;

  // gather embeddings directly into A-fragment order: lane covers (seq, K-halves)
  const int* xrow = x + (tile * 16 + seq) * 16;
  unsigned   mw   = 0;
#pragma unroll
  for (int t = 0; t < 16; ++t) {
    int tok = xrow[t];
    mw |= (tok != 0) ? (1u << t) : 0u;
    const float* er = etab + tok * 32 + k0;
    float4 e0 = *(const float4*)(er);
    float4 e1 = *(const float4*)(er + 4);
    float4 e2 = *(const float4*)(er + 16);
    float4 e3 = *(const float4*)(er + 20);
    v8h p0 = { (_Float16)e0.x, (_Float16)e0.y, (_Float16)e0.z, (_Float16)e0.w,
               (_Float16)e1.x, (_Float16)e1.y, (_Float16)e1.z, (_Float16)e1.w };
    v8h p1 = { (_Float16)e2.x, (_Float16)e2.y, (_Float16)e2.z, (_Float16)e2.w,
               (_Float16)e3.x, (_Float16)e3.y, (_Float16)e3.z, (_Float16)e3.w };
    _Float16* d = &ebuf[w][(t * 32 + lane) * 16];
    *(v8h*)d       = p0;
    *(v8h*)(d + 8) = p1;
  }
  if (!hi) mbuf[w][seq] = mw;
  __syncthreads();

  unsigned um[8];
#pragma unroll
  for (int r = 0; r < 8; ++r) um[r] = mbuf[w][r + (hi ? 8 : 0)];
  float bv[8];
#pragma unroll
  for (int j = 0; j < 8; ++j) bv[j] = bias[j * 16 + (lane & 15)];

  _Float16* st = state + (size_t)tile * 16 * 64;
  run_dir<true >(ebuf[w], wkf, wrf, hbuf[w], um, bv, lane, st);        // h_fw -> cols 0..31
  run_dir<false>(ebuf[w], wkf, wrf, hbuf[w], um, bv, lane, st + 32);   // h_bw -> cols 32..63
}

// ---------------------------------------------------------------------------
// Kernel 2: out = tanh(state[2048,4096](f16) @ WdF + bd).  1 wave == 16 boards,
// all 4 column chunks; 128 K-steps of 32.  128 tiles -> 32 blocks x 128 thr.
// ---------------------------------------------------------------------------
__global__ __launch_bounds__(128) void head_kernel(const _Float16* __restrict__ state,
                                                   const _Float16* __restrict__ wdf,
                                                   const float* __restrict__ bd,
                                                   float* __restrict__ out) {
  const int  tid  = threadIdx.x;
  const int  w    = tid >> 5;
  const int  lane = tid & 31;
  const int  tile = blockIdx.x * 4 + w;
  const int  col  = lane & 15;
  const bool hi   = lane >= 16;
  const int  k0   = hi ? 8 : 0;

  v8f acc[4];
#pragma unroll
  for (int j = 0; j < 4; ++j) {
    float b = bd[j * 16 + col];
#pragma unroll
    for (int r = 0; r < 8; ++r) acc[j][r] = b;
  }

  const _Float16* arow = state + (size_t)(tile * 16 + col) * 4096;
  for (int s = 0; s < 128; ++s) {
    const _Float16* ap = arow + 32 * s + k0;
    v16h af = cat8(*(const v8h*)ap, *(const v8h*)(ap + 16));
#pragma unroll
    for (int j = 0; j < 4; ++j) {
      const _Float16* bp = wdf + ((s * 4 + j) * 32 + lane) * 16;
      v16h bf = cat8(*(const v8h*)bp, *(const v8h*)(bp + 8));
      acc[j] = __builtin_amdgcn_wmma_f32_16x16x32_f16(false, af, false, bf, (short)0, acc[j], false, false);
    }
  }

#pragma unroll
  for (int j = 0; j < 4; ++j)
#pragma unroll
    for (int r = 0; r < 8; ++r) {
      int row = tile * 16 + r + (hi ? 8 : 0);
      out[row * 64 + j * 16 + col] = ftanh(acc[j][r]);
    }
}

// ---------------------------------------------------------------------------
extern "C" void kernel_launch(void* const* d_in, const int* in_sizes, int n_in,
                              void* d_out, int out_size, void* d_ws, size_t ws_size,
                              hipStream_t stream) {
  const int*   x    = (const int*)d_in[0];
  const float* etab = (const float*)d_in[1];
  const float* Wk   = (const float*)d_in[2];
  const float* Wr   = (const float*)d_in[3];
  const float* b    = (const float*)d_in[4];
  const float* Wd   = (const float*)d_in[5];
  const float* bd   = (const float*)d_in[6];
  float*       out  = (float*)d_out;

  char*      ws    = (char*)d_ws;                      // needs ~17.1 MB
  _Float16*  wkf   = (_Float16*)(ws + WKF_OFF);
  _Float16*  wrf   = (_Float16*)(ws + WRF_OFF);
  _Float16*  wdf   = (_Float16*)(ws + WDF_OFF);
  _Float16*  state = (_Float16*)(ws + STATE_OFF);      // [131072][64] f16

  prep_kernel<<<1056, 256, 0, stream>>>(Wk, Wr, Wd, wkf, wrf, wdf);
  lstm_kernel<<<4096, 64, 0, stream>>>(x, etab, b, wkf, wrf, state);
  head_kernel<<<32, 128, 0, stream>>>(state, wdf, bd, out);
}
